// Mask3dAttention_66297115181710
// MI455X (gfx1250) — matrix operand order repaired and checked by reading, not compiled, not run on hardware
//
#include <hip/hip_runtime.h>

// ---------------------------------------------------------------------------
// Mask3dAttention for MI455X (gfx1250): fused QKV GEMM + flash attention.
// Matmuls on v_wmma_f32_16x16x32_bf16 (wave32 WMMA), f32 accumulation.
// K/V tiles staged by the Tensor Data Mover (tensor_load_to_lds, TENSORcnt),
// double-buffered so DMA overlaps WMMA + online softmax.
// ---------------------------------------------------------------------------

#define BATCH   4
#define SEQ     2048
#define DIM     256
#define NHEAD   8
#define HD      32
#define N3      768           // 3*DIM
#define HALFSEQ 1024
#define SCALE_Q 0.17677669529663687f   // 32^-0.5
#define MASKNEG (-1000.0f)

typedef __attribute__((ext_vector_type(16))) __bf16 v16bf;
typedef __attribute__((ext_vector_type(8)))  float  v8f;
typedef __attribute__((ext_vector_type(4)))  unsigned int v4u;
typedef __attribute__((ext_vector_type(8)))  int    v8i;
typedef __attribute__((ext_vector_type(4)))  int    v4i;

#if defined(__gfx1250__) && __has_builtin(__builtin_amdgcn_tensor_load_to_lds) && \
    __has_builtin(__builtin_amdgcn_s_wait_tensorcnt)
#define HAVE_TDM 1
#else
#define HAVE_TDM 0
#endif

union ABFrag {
    v16bf v;
    unsigned short s[16];
    unsigned int   u[8];
};

// f32 -> bf16 round-to-nearest-even
__device__ __forceinline__ unsigned short f2bf(float f) {
    union { float f; unsigned int u; } c;
    c.f = f;
    unsigned int r = c.u + 0x7FFFu + ((c.u >> 16) & 1u);
    return (unsigned short)(r >> 16);
}

// Low 32 bits of a generic shared pointer are the LDS byte offset
// (generic shared addr = {SHARED_BASE, lds_offset}, ISA 10.2 aperture rules).
__device__ __forceinline__ unsigned lds_byte_addr(const void* p) {
    return (unsigned)(unsigned long long)p;
}

#if HAVE_TDM
// Issue one 2D TDM load: tile 32 x 32 bf16 (tile_dim0=32 contiguous elements
// per row, tile_dim1=32 rows), LDS padded +1 DWORD every 16 DWORDs so the
// destination row stride is 68B (34 bf16) - bank-conflict-free for frag reads.
__device__ __forceinline__ void tdm_load_tile32x32(unsigned lds_addr,
                                                   const void* gptr,
                                                   unsigned tensor_dim0,
                                                   unsigned tensor_dim1,
                                                   unsigned stride0) {
    const unsigned long long ga = (unsigned long long)gptr;
    v4u g0;
    g0[0] = 1u;                                              // count=1 (valid)
    g0[1] = lds_addr;                                        // lds_addr[31:0]
    g0[2] = (unsigned)ga;                                    // global_addr lo
    g0[3] = (unsigned)((ga >> 32) & 0x01FFFFFFu) | (2u << 30); // addr hi | type=2
    v8i g1;
    // data_size=1 (2B), pad_enable=1, pad_interval=3 (16 DW), pad_amount=0 (1 DW)
    g1[0] = (int)((1u << 16) | (1u << 20) | (3u << 22));
    g1[1] = (int)(tensor_dim0 << 16);                        // tensor_dim0[15:0]
    g1[2] = (int)(((tensor_dim1 & 0xFFFFu) << 16) | (tensor_dim0 >> 16));
    g1[3] = (int)((32u << 16) | (tensor_dim1 >> 16));        // tile_dim0=32
    g1[4] = 32;                                              // tile_dim1=32
    g1[5] = (int)stride0;                                    // dim0 stride lo32
    g1[6] = 0;
    g1[7] = 0;
    v4i z4 = {0, 0, 0, 0};
#if __clang_major__ >= 23
    v8i z8 = {0, 0, 0, 0, 0, 0, 0, 0};
    __builtin_amdgcn_tensor_load_to_lds(g0, g1, z4, z4, z8, 0);
#else
    __builtin_amdgcn_tensor_load_to_lds(g0, g1, z4, z4, 0);
#endif
}
#endif // HAVE_TDM

// ===========================================================================
// Kernel 1: qkv = x @ W_qkv + b_qkv ; emit Q*scale, K as [BH, N, 32] bf16,
// V transposed as [BH, 32, N] bf16 (so attention B-fragments read contiguous).
// Block = 256 threads = 8 waves; tile 128(M) x 64(Nout); K-step 32.
// ===========================================================================
#define MT 128
#define NT 64
#define KT 32
#define A_STRIDE 34    // bf16 elements (68B rows, 4B aligned, conflict-free)
#define B_STRIDE 34

__global__ __launch_bounds__(256)
void qkv_gemm_bf16_wmma(const float* __restrict__ x,
                        const float* __restrict__ W,
                        const float* __restrict__ bias,
                        unsigned short* __restrict__ Qb,
                        unsigned short* __restrict__ Kb,
                        unsigned short* __restrict__ Vt)
{
    __shared__ unsigned short As[MT * A_STRIDE];   // [row][k]   (x tile, bf16)
    __shared__ unsigned short Bs[NT * B_STRIDE];   // [n][k]     (W^T tile, bf16)

    const int t     = threadIdx.x;
    const int lane  = t & 31;
    const int wave  = t >> 5;
    const int half  = lane >> 4;
    const int l16   = lane & 15;
    const int mtile = blockIdx.x * MT;
    const int ntile = blockIdx.y * NT;

    v8f acc[4] = {};   // 16x64 per wave = 4 C fragments

    for (int kk = 0; kk < DIM; kk += KT) {
        __syncthreads();
        // ---- stage A tile: 128x32 f32 -> bf16 (each thread: 16 elements)
        {
            const int row = t >> 1, c0 = (t & 1) * 16;
            const float* src = x + (size_t)(mtile + row) * DIM + kk + c0;
            unsigned short* dst = As + row * A_STRIDE + c0;
            #pragma unroll
            for (int i = 0; i < 16; ++i) dst[i] = f2bf(src[i]);
        }
        // ---- stage B tile transposed: Bs[n][k] = W[kk+k][ntile+n]
        {
            const int k = t >> 3, n0 = (t & 7) * 8;
            const float* src = W + (size_t)(kk + k) * N3 + ntile + n0;
            #pragma unroll
            for (int i = 0; i < 8; ++i) Bs[(n0 + i) * B_STRIDE + k] = f2bf(src[i]);
        }
        __syncthreads();

        // ---- A fragment (16x32): lane holds row l16, K = half*16 + 0..15
        ABFrag a;
        {
            const unsigned short* base = As + (wave * 16 + l16) * A_STRIDE + half * 16;
            #pragma unroll
            for (int v = 0; v < 8; ++v) {
                const int kb = 2 * v;
                a.u[v] = *(const unsigned int*)(base + kb);
            }
        }
        // ---- 4 x (B fragment + WMMA)
        #pragma unroll
        for (int nt = 0; nt < 4; ++nt) {
            ABFrag bf;
            const unsigned short* base = Bs + (nt * 16 + l16) * B_STRIDE + half * 16;
            #pragma unroll
            for (int v = 0; v < 8; ++v) bf.u[v] = *(const unsigned int*)(base + 2 * v);
            acc[nt] = __builtin_amdgcn_wmma_f32_16x16x32_bf16(
                false, a.v, false, bf.v, (short)0, acc[nt], false, false);
        }
    }

    // ---- writeout: split qkv columns into Q (scaled), K, V^T
    #pragma unroll
    for (int nt = 0; nt < 4; ++nt) {
        const int c  = ntile + nt * 16 + l16;   // 0..767
        const float bv = bias[c];
        const int sel = c >> 8;                 // 0=q 1=k 2=v
        const int cc  = c & 255;
        const int h   = cc >> 5, d = cc & 31;
        #pragma unroll
        for (int r = 0; r < 8; ++r) {
            const int m = mtile + wave * 16 + r + half * 8;   // row of x (B*N flat)
            const int b = m >> 11, n = m & (SEQ - 1);
            const float val = acc[nt][r] + bv;
            const size_t bh = (size_t)(b * NHEAD + h);
            if (sel == 0)
                Qb[(bh * SEQ + n) * HD + d] = f2bf(val * SCALE_Q);
            else if (sel == 1)
                Kb[(bh * SEQ + n) * HD + d] = f2bf(val);
            else
                Vt[(bh * HD + d) * SEQ + n] = f2bf(val);
        }
    }
}

// ===========================================================================
// Kernel 2: flash attention. grid = (N/64, B*H), block = 128 (4 waves).
// Each wave owns 16 query rows; loops over 2048 keys in tiles of 32.
// K/V tiles staged by TDM into double-buffered LDS (68B padded rows).
// ===========================================================================
#define KTILE 32
#define KV_STRIDE 34   // bf16 elems = 68B rows (matches TDM pad of 1DW/16DW)
#define P_STRIDE  34

__global__ __launch_bounds__(128)
void flash_attn_bf16_wmma(const unsigned short* __restrict__ Qb,
                          const unsigned short* __restrict__ Kb,
                          const unsigned short* __restrict__ Vt,
                          float* __restrict__ out)
{
    __shared__ unsigned short Ks[2][KTILE * KV_STRIDE];   // [key][d]
    __shared__ unsigned short Vs[2][HD    * KV_STRIDE];   // [d][key]  (V^T tile)
    __shared__ unsigned short Ps[4][16 * P_STRIDE];       // per-wave P scratch

    const int t    = threadIdx.x;
    const int lane = t & 31;
    const int wave = t >> 5;
    const int half = lane >> 4;
    const int l16  = lane & 15;
    const int q0   = blockIdx.x * 64 + wave * 16;
    const int bh   = blockIdx.y;                          // b*8 + h

    const unsigned short* Qbh = Qb + (size_t)bh * SEQ * HD;
    const unsigned short* Kbh = Kb + (size_t)bh * SEQ * HD;
    const unsigned short* Vbh = Vt + (size_t)bh * HD * SEQ;

    // Q A-fragment, held in registers for the whole key loop
    ABFrag qf;
    {
        const unsigned short* base = Qbh + (size_t)(q0 + l16) * HD + half * 16;
        #pragma unroll
        for (int v = 0; v < 8; ++v) {
            const int db = 2 * v;
            qf.u[v] = *(const unsigned int*)(base + db);
        }
    }

    float Mrow[8], Lrow[8];
    v8f o0 = {}, o1 = {};
    #pragma unroll
    for (int r = 0; r < 8; ++r) { Mrow[r] = -1e30f; Lrow[r] = 0.0f; }

#if HAVE_TDM
    // Prologue: DMA first K/V tile into buffer 0 (one wave issues; EXEC ignored)
    if (wave == 0) {
        tdm_load_tile32x32(lds_byte_addr(Ks[0]), Kbh, HD, SEQ, HD);
        tdm_load_tile32x32(lds_byte_addr(Vs[0]), Vbh, SEQ, HD, SEQ);
    }
#endif

    for (int kt = 0; kt < SEQ / KTILE; ++kt) {
        const int k0  = kt * KTILE;
        const int cur = kt & 1;

        __syncthreads();   // all waves done with buffer !cur -> reusable
#if HAVE_TDM
        if (wave == 0) {
            if (kt + 1 < SEQ / KTILE) {
                // pipeline: issue next tile's DMA, then wait only for current
                tdm_load_tile32x32(lds_byte_addr(Ks[1 - cur]),
                                   Kbh + (size_t)(k0 + KTILE) * HD, HD, SEQ, HD);
                tdm_load_tile32x32(lds_byte_addr(Vs[1 - cur]),
                                   Vbh + (size_t)(k0 + KTILE), SEQ, HD, SEQ);
                __builtin_amdgcn_s_wait_tensorcnt(2);   // current 2 loads done
            } else {
                __builtin_amdgcn_s_wait_tensorcnt(0);
            }
        }
#else
        // Fallback: cooperative VGPR staging (u32 granularity, 4B aligned)
        {
            const int row = t >> 2;            // 0..31
            const int c0  = (t & 3) * 8;       // 8 bf16 = 4 u32 per thread/row
            const unsigned int* ks = (const unsigned int*)(Kbh + (size_t)(k0 + row) * HD + c0);
            unsigned int* kd = (unsigned int*)(Ks[cur] + row * KV_STRIDE + c0);
            #pragma unroll
            for (int i = 0; i < 4; ++i) kd[i] = ks[i];
            const unsigned int* vs = (const unsigned int*)(Vbh + (size_t)row * SEQ + k0 + c0);
            unsigned int* vd = (unsigned int*)(Vs[cur] + row * KV_STRIDE + c0);
            #pragma unroll
            for (int i = 0; i < 4; ++i) vd[i] = vs[i];
        }
        if (k0 + KTILE < SEQ) {   // warm next tile (global_prefetch_b8)
            __builtin_prefetch(Kbh + (size_t)(k0 + KTILE + (t >> 2)) * HD, 0, 1);
            __builtin_prefetch(Vbh + (size_t)(t >> 2) * SEQ + k0 + KTILE, 0, 1);
        }
#endif
        __syncthreads();   // current buffer published to all waves

        // ---- S = Q K^T : two 16(q) x 16(key) tiles (K-dim = head_dim = 32)
        v8f s0 = {}, s1 = {};
        {
            ABFrag kb;
            const unsigned short* b0 = Ks[cur] + l16 * KV_STRIDE + half * 16;
            #pragma unroll
            for (int v = 0; v < 8; ++v) kb.u[v] = *(const unsigned int*)(b0 + 2 * v);
            s0 = __builtin_amdgcn_wmma_f32_16x16x32_bf16(
                false, qf.v, false, kb.v, (short)0, s0, false, false);
            const unsigned short* b1 = Ks[cur] + (16 + l16) * KV_STRIDE + half * 16;
            #pragma unroll
            for (int v = 0; v < 8; ++v) kb.u[v] = *(const unsigned int*)(b1 + 2 * v);
            s1 = __builtin_amdgcn_wmma_f32_16x16x32_bf16(
                false, qf.v, false, kb.v, (short)0, s1, false, false);
        }

        // ---- mask + online softmax (rows live in one 16-lane half)
        const int keyA = k0 + l16;
        const int keyB = k0 + 16 + l16;
        #pragma unroll
        for (int r = 0; r < 8; ++r) {
            const int q = q0 + r + half * 8;
            const bool visA = (q < HALFSEQ) ? (keyA < HALFSEQ) : (keyA <= q);
            const bool visB = (q < HALFSEQ) ? (keyB < HALFSEQ) : (keyB <= q);
            const float a  = s0[r] + (visA ? 0.0f : MASKNEG);
            const float b2 = s1[r] + (visB ? 0.0f : MASKNEG);

            float mx = fmaxf(a, b2);
            #pragma unroll
            for (int off = 1; off < 16; off <<= 1)
                mx = fmaxf(mx, __shfl_xor(mx, off, 32));
            const float nm   = fmaxf(Mrow[r], mx);
            const float corr = __expf(Mrow[r] - nm);
            const float pa   = __expf(a  - nm);
            const float pb   = __expf(b2 - nm);
            float rs = pa + pb;
            #pragma unroll
            for (int off = 1; off < 16; off <<= 1)
                rs += __shfl_xor(rs, off, 32);
            Lrow[r] = Lrow[r] * corr + rs;
            Mrow[r] = nm;
            o0[r] *= corr;
            o1[r] *= corr;
            // spill P (C layout) to wave-private LDS for re-read in A layout;
            // DS ops are in-order within a wave, no barrier required.
            unsigned short* pr = Ps[wave] + (r + half * 8) * P_STRIDE;
            pr[l16]      = f2bf(pa);
            pr[16 + l16] = f2bf(pb);
        }

        // ---- reload P as A-fragment (16 q x 32 key)
        ABFrag pf;
        {
            const unsigned short* base = Ps[wave] + l16 * P_STRIDE + half * 16;
            #pragma unroll
            for (int v = 0; v < 8; ++v) {
                const int kb = 2 * v;
                pf.u[v] = *(const unsigned int*)(base + kb);
            }
        }
        // ---- O += P V  (two d tiles), accumulate via WMMA C operand
        {
            ABFrag vb;
            const unsigned short* b0 = Vs[cur] + l16 * KV_STRIDE + half * 16;
            #pragma unroll
            for (int v = 0; v < 8; ++v) vb.u[v] = *(const unsigned int*)(b0 + 2 * v);
            o0 = __builtin_amdgcn_wmma_f32_16x16x32_bf16(
                false, pf.v, false, vb.v, (short)0, o0, false, false);
            const unsigned short* b1 = Vs[cur] + (16 + l16) * KV_STRIDE + half * 16;
            #pragma unroll
            for (int v = 0; v < 8; ++v) vb.u[v] = *(const unsigned int*)(b1 + 2 * v);
            o1 = __builtin_amdgcn_wmma_f32_16x16x32_bf16(
                false, pf.v, false, vb.v, (short)0, o1, false, false);
        }
    }

    // ---- normalize + writeout: out[b][q][h*32 + d], f32
    const int b = bh >> 3, h = bh & 7;
    #pragma unroll
    for (int r = 0; r < 8; ++r) {
        const int q = q0 + r + half * 8;
        const float inv = 1.0f / Lrow[r];
        float* dst = out + ((size_t)(b * SEQ + q)) * DIM + h * HD;
        dst[l16]      = o0[r] * inv;
        dst[16 + l16] = o1[r] * inv;
    }
}

// ===========================================================================
extern "C" void kernel_launch(void* const* d_in, const int* in_sizes, int n_in,
                              void* d_out, int out_size, void* d_ws, size_t ws_size,
                              hipStream_t stream) {
    const float* x    = (const float*)d_in[0];   // [4, 2048, 256]
    const float* Wqkv = (const float*)d_in[1];   // [256, 768]
    const float* bqkv = (const float*)d_in[2];   // [768]

    const size_t perTensor = (size_t)BATCH * NHEAD * SEQ * HD;  // 2,097,152 bf16
    unsigned short* Qb = (unsigned short*)d_ws;                 // [BH, N, 32]
    unsigned short* Kb = Qb + perTensor;                        // [BH, N, 32]
    unsigned short* Vt = Kb + perTensor;                        // [BH, 32, N]

    dim3 g1((BATCH * SEQ) / MT, N3 / NT);        // 64 x 12
    qkv_gemm_bf16_wmma<<<g1, 256, 0, stream>>>(x, Wqkv, bqkv, Qb, Kb, Vt);

    dim3 g2(SEQ / 64, BATCH * NHEAD);            // 32 x 32
    flash_attn_bf16_wmma<<<g2, 128, 0, stream>>>(Qb, Kb, Vt, (float*)d_out);
}
